// LstmModel_29892972380168
// MI455X (gfx1250) — matrix operand order repaired and checked by reading, compiled, not run on hardware
//
#include <hip/hip_runtime.h>

// Problem constants (match reference).
#define T_STEPS 256
#define BATCH   64
#define LAYERS  2
#define HID     1024
#define INPUT   1024
#define GATES   4096   // 4*HID, PyTorch gate order i,f,g,o
#define CHUNK   64     // timesteps of xg precomputed per GEMM launch

typedef __attribute__((ext_vector_type(16))) __bf16 v16bf;
typedef __attribute__((ext_vector_type(8)))  __bf16 v8bf;
typedef __attribute__((ext_vector_type(8)))  float  v8f;

__device__ __forceinline__ __bf16 f2bf(float f) {
  unsigned u = __builtin_bit_cast(unsigned, f);
  unsigned r = u + 0x7FFFu + ((u >> 16) & 1u);   // round-to-nearest-even
  unsigned short h = (unsigned short)(r >> 16);
  return __builtin_bit_cast(__bf16, h);
}

__device__ __forceinline__ float sigm(float x) { return 1.0f / (1.0f + __expf(-x)); }

// ---------------------------------------------------------------------------
// Elementwise f32 -> bf16 conversion (grid-stride).
// ---------------------------------------------------------------------------
__global__ __launch_bounds__(256) void f32_to_bf16_kernel(
    const float* __restrict__ src, __bf16* __restrict__ dst, int n) {
  int i = blockIdx.x * blockDim.x + threadIdx.x;
  int stride = gridDim.x * blockDim.x;
  for (; i < n; i += stride) dst[i] = f2bf(src[i]);
}

// ---------------------------------------------------------------------------
// Per-layer state init: h (bf16) and c (f32) from [B, L, H] inputs.
// ---------------------------------------------------------------------------
__global__ __launch_bounds__(256) void init_state_kernel(
    const float* __restrict__ h0, const float* __restrict__ c0, int layer,
    __bf16* __restrict__ hbuf, float* __restrict__ cbuf) {
  int i = blockIdx.x * blockDim.x + threadIdx.x;   // over B*H
  if (i >= BATCH * HID) return;
  int b = i / HID, h = i % HID;
  size_t src = ((size_t)b * LAYERS + layer) * HID + h;
  hbuf[i] = f2bf(h0[src]);
  cbuf[i] = c0[src];
}

// ---------------------------------------------------------------------------
// Input projection GEMM: xg[r, g] = sum_k A[r,k] * W[g,k] + (bi[g]+bh[g])
//   A: [CHUNK*BATCH, 1024] bf16 row-major, W: [GATES, 1024] bf16 row-major.
// Workgroup = 8 waves covering a 128x128 tile; each wave computes 64x32
// via 4x2 accumulator tiles of v_wmma_f32_16x16x32_bf16.
// A frag (16x32): lanes 0-15 hold M=lane, K 0..15; lanes 16-31 hold K 16..31.
// B frag (32x16): lane holds column n=lane&15, 16 contiguous K (= a row slice of W).
// ---------------------------------------------------------------------------
__global__ __launch_bounds__(256) void xg_gemm_kernel(
    const __bf16* __restrict__ A,
    const __bf16* __restrict__ W,
    const float*  __restrict__ bi,
    const float*  __restrict__ bh,
    float* __restrict__ xg) {
  const int K = 1024;
  const int lane = threadIdx.x & 31;
  const int wave = threadIdx.x >> 5;
  const int mBase = blockIdx.x * 128 + (wave & 1) * 64;
  const int nBase = blockIdx.y * 128 + (wave >> 1) * 32;
  const int mrow  = lane & 15;
  const int half  = lane >> 4;

  v8f acc[4][2] = {};
  for (int k = 0; k < K; k += 32) {
    v16bf a[4], b[2];
#pragma unroll
    for (int mt = 0; mt < 4; ++mt) {
      const __bf16* p = A + (size_t)(mBase + mt * 16 + mrow) * K + k + half * 16;
      v8bf lo = *reinterpret_cast<const v8bf*>(p);
      v8bf hi = *reinterpret_cast<const v8bf*>(p + 8);
      a[mt] = __builtin_shufflevector(lo, hi, 0, 1, 2, 3, 4, 5, 6, 7,
                                      8, 9, 10, 11, 12, 13, 14, 15);
    }
#pragma unroll
    for (int nt = 0; nt < 2; ++nt) {
      const __bf16* p = W + (size_t)(nBase + nt * 16 + mrow) * K + k + half * 16;
      b[nt] = *reinterpret_cast<const v16bf*>(p);
    }
#pragma unroll
    for (int mt = 0; mt < 4; ++mt)
#pragma unroll
      for (int nt = 0; nt < 2; ++nt)
        acc[mt][nt] = __builtin_amdgcn_wmma_f32_16x16x32_bf16(
            false, a[mt], false, b[nt], (short)0, acc[mt][nt], false, false);
  }

#pragma unroll
  for (int nt = 0; nt < 2; ++nt) {
    const int col = nBase + nt * 16 + mrow;
    const float bsum = bi[col] + bh[col];
#pragma unroll
    for (int mt = 0; mt < 4; ++mt) {
#pragma unroll
      for (int v = 0; v < 8; ++v) {
        const int row = mBase + mt * 16 + half * 8 + v;
        xg[(size_t)row * GATES + col] = acc[mt][nt][v] + bsum;
      }
    }
  }
}

// ---------------------------------------------------------------------------
// One LSTM timestep: gates = xg_t + h_prev @ Whh^T, then fused cell update.
// Grid: 32 workgroups (each owns 32 hidden columns), 256 threads = 8 waves.
// Wave w computes gate (w>>1) for 16 of the 32 columns: a 64x16 GEMM slab
// (4 M-tiles, K=1024 in steps of 32). Accumulators -> LDS, barrier, then
// fp32 gate math over the workgroup's 64x32 patch.
// ---------------------------------------------------------------------------
__global__ __launch_bounds__(256) void lstm_step_kernel(
    const __bf16* __restrict__ hprev,  // [B, H] bf16
    const __bf16* __restrict__ Whh,    // [GATES, H] bf16 row-major
    const float*  __restrict__ xg_t,   // [B, GATES] slice for this timestep
    float*  __restrict__ cbuf,         // [B, H] cell state (in/out)
    __bf16* __restrict__ hnext,        // [B, H] bf16 (next step's input)
    __bf16* __restrict__ ybf,          // layer-0: bf16 out row [B, H]; else null
    float*  __restrict__ yf,           // layer-1: f32 out row [B, H]; else null
    float*  __restrict__ hn,           // d_out hn region [B, L, H]
    float*  __restrict__ cn,           // d_out cn region [B, L, H]
    int layer, int writeState) {
  __shared__ float gl[4][BATCH][32];

  const int lane = threadIdx.x & 31;
  const int wave = threadIdx.x >> 5;
  const int gate = wave >> 1;
  const int nLoc = (wave & 1) * 16;                 // column offset within WG's 32
  const int gcol = gate * HID + blockIdx.x * 32 + nLoc;  // column in gate space
  const int mrow = lane & 15;
  const int half = lane >> 4;

  v8f acc[4] = {};
  for (int k = 0; k < HID; k += 32) {
    v16bf a[4];
#pragma unroll
    for (int mt = 0; mt < 4; ++mt) {
      const __bf16* p = hprev + (size_t)(mt * 16 + mrow) * HID + k + half * 16;
      v8bf lo = *reinterpret_cast<const v8bf*>(p);
      v8bf hi = *reinterpret_cast<const v8bf*>(p + 8);
      a[mt] = __builtin_shufflevector(lo, hi, 0, 1, 2, 3, 4, 5, 6, 7,
                                      8, 9, 10, 11, 12, 13, 14, 15);
    }
    const __bf16* pb = Whh + (size_t)(gcol + mrow) * HID + k + half * 16;
    v16bf b = *reinterpret_cast<const v16bf*>(pb);
#pragma unroll
    for (int mt = 0; mt < 4; ++mt)
      acc[mt] = __builtin_amdgcn_wmma_f32_16x16x32_bf16(
          false, a[mt], false, b, (short)0, acc[mt], false, false);
  }

#pragma unroll
  for (int mt = 0; mt < 4; ++mt)
#pragma unroll
    for (int v = 0; v < 8; ++v)
      gl[gate][mt * 16 + half * 8 + v][nLoc + mrow] = acc[mt][v];

  __syncthreads();

  // Fused gate nonlinearity over the 64x32 patch (8 elements per thread).
  for (int e = threadIdx.x; e < BATCH * 32; e += 256) {
    const int m = e >> 5;
    const int j = e & 31;
    const int hcol = blockIdx.x * 32 + j;
    const size_t xr = (size_t)m * GATES;
    float gi = gl[0][m][j] + xg_t[xr + 0 * HID + hcol];
    float gf = gl[1][m][j] + xg_t[xr + 1 * HID + hcol];
    float gg = gl[2][m][j] + xg_t[xr + 2 * HID + hcol];
    float go = gl[3][m][j] + xg_t[xr + 3 * HID + hcol];
    const size_t ci = (size_t)m * HID + hcol;
    float cNew = sigm(gf) * cbuf[ci] + sigm(gi) * tanhf(gg);
    float hNew = sigm(go) * tanhf(cNew);
    cbuf[ci]  = cNew;
    hnext[ci] = f2bf(hNew);
    if (ybf) ybf[ci] = f2bf(hNew);
    if (yf)  yf[ci]  = hNew;
    if (writeState) {
      const size_t si = ((size_t)m * LAYERS + layer) * HID + hcol;
      hn[si] = hNew;
      cn[si] = cNew;
    }
  }
}

// ---------------------------------------------------------------------------
// Host launch
// ---------------------------------------------------------------------------
extern "C" void kernel_launch(void* const* d_in, const int* in_sizes, int n_in,
                              void* d_out, int out_size, void* d_ws, size_t ws_size,
                              hipStream_t stream) {
  const float* x   = (const float*)d_in[0];  // [T, B, IN]
  const float* h0  = (const float*)d_in[1];  // [B, L, H]
  const float* c0  = (const float*)d_in[2];  // [B, L, H]
  const float* Wih = (const float*)d_in[3];  // [L, 4H, IN]
  const float* Whh = (const float*)d_in[4];  // [L, 4H, H]
  const float* bih = (const float*)d_in[5];  // [L, 4H]
  const float* bhh = (const float*)d_in[6];  // [L, 4H]

  float* out = (float*)d_out;                         // [T, B, H]
  float* hn  = out + (size_t)T_STEPS * BATCH * HID;   // [B, L, H]
  float* cn  = hn + (size_t)BATCH * LAYERS * HID;     // [B, L, H]

  // Workspace carve-up (~161 MB).
  char* p = (char*)d_ws;
  __bf16* x_bf   = (__bf16*)p; p += (size_t)T_STEPS * BATCH * INPUT * 2;
  __bf16* y0_bf  = (__bf16*)p; p += (size_t)T_STEPS * BATCH * HID * 2;
  __bf16* Wih_bf = (__bf16*)p; p += (size_t)LAYERS * GATES * INPUT * 2;
  __bf16* Whh_bf = (__bf16*)p; p += (size_t)LAYERS * GATES * HID * 2;
  float*  xg     = (float*)p;  p += (size_t)CHUNK * BATCH * GATES * 4;
  __bf16* hbuf0  = (__bf16*)p; p += (size_t)BATCH * HID * 2;
  __bf16* hbuf1  = (__bf16*)p; p += (size_t)BATCH * HID * 2;
  float*  cbuf   = (float*)p;  p += (size_t)BATCH * HID * 4;

  // Precision down-conversion of GEMM operands.
  {
    int n = T_STEPS * BATCH * INPUT;
    f32_to_bf16_kernel<<<2048, 256, 0, stream>>>(x, x_bf, n);
    n = LAYERS * GATES * INPUT;
    f32_to_bf16_kernel<<<2048, 256, 0, stream>>>(Wih, Wih_bf, n);
    n = LAYERS * GATES * HID;
    f32_to_bf16_kernel<<<2048, 256, 0, stream>>>(Whh, Whh_bf, n);
  }

  const int nChunks = T_STEPS / CHUNK;

  for (int l = 0; l < LAYERS; ++l) {
    init_state_kernel<<<(BATCH * HID + 255) / 256, 256, 0, stream>>>(
        h0, c0, l, hbuf0, cbuf);

    const __bf16* Ain  = (l == 0) ? x_bf : y0_bf;
    const __bf16* Wl   = Wih_bf + (size_t)l * GATES * INPUT;
    const __bf16* Whl  = Whh_bf + (size_t)l * GATES * HID;
    const float*  bil  = bih + (size_t)l * GATES;
    const float*  bhl  = bhh + (size_t)l * GATES;

    for (int ch = 0; ch < nChunks; ++ch) {
      // Batched input projection for CHUNK timesteps: [CHUNK*B, 1024]x[1024, 4096].
      xg_gemm_kernel<<<dim3(CHUNK * BATCH / 128, GATES / 128), 256, 0, stream>>>(
          Ain + (size_t)ch * CHUNK * BATCH * 1024, Wl, bil, bhl, xg);

      for (int tc = 0; tc < CHUNK; ++tc) {
        const int t = ch * CHUNK + tc;
        __bf16* hr = (t & 1) ? hbuf1 : hbuf0;
        __bf16* hw = (t & 1) ? hbuf0 : hbuf1;
        const float* xg_t = xg + (size_t)tc * BATCH * GATES;
        __bf16* ybf = (l == 0) ? (y0_bf + (size_t)t * BATCH * HID) : nullptr;
        float*  yf  = (l == 1) ? (out + (size_t)t * BATCH * HID) : nullptr;
        lstm_step_kernel<<<HID / 32, 256, 0, stream>>>(
            hr, Whl, xg_t, cbuf, hw, ybf, yf, hn, cn, l,
            (t == T_STEPS - 1) ? 1 : 0);
      }
    }
  }
}
